// CosineAdjConstruction_35759897706677
// MI455X (gfx1250) — compile-verified, matrix operand order checked by reading, not run on hardware
//
#include <hip/hip_runtime.h>
#include <hip/hip_bf16.h>

typedef __attribute__((ext_vector_type(16))) _Float16 v16h;
typedef __attribute__((ext_vector_type(8)))  _Float16 v8h;
typedef __attribute__((ext_vector_type(8)))  float    v8f;

#define N_DIM 4096
#define D_DIM 512
#define TILE  128
#define KC    64
#define LDK   72                 // padded LDS row stride in halves (144 B)
#define TILE_HALVES (TILE * LDK) // one tile buffer in halves
#define NCHUNK (D_DIM / KC)      // 8

// ---------------------------------------------------------------------------
// Kernel 1: row-normalize x (fp32) -> xn (f16). One wave32 per 512-elem row.
// ---------------------------------------------------------------------------
__global__ __launch_bounds__(256) void normalize_kernel(
    const float* __restrict__ x, _Float16* __restrict__ xn) {
  const int lane = threadIdx.x & 31;
  const int row  = blockIdx.x * 8 + (threadIdx.x >> 5);   // 8 waves per block

  const float* xr = x + (size_t)row * D_DIM + lane * 16;
  float4 f0 = ((const float4*)xr)[0];
  float4 f1 = ((const float4*)xr)[1];
  float4 f2 = ((const float4*)xr)[2];
  float4 f3 = ((const float4*)xr)[3];

  float v[16] = {f0.x, f0.y, f0.z, f0.w, f1.x, f1.y, f1.z, f1.w,
                 f2.x, f2.y, f2.z, f2.w, f3.x, f3.y, f3.z, f3.w};
  float ss = 0.f;
#pragma unroll
  for (int i = 0; i < 16; ++i) ss += v[i] * v[i];
#pragma unroll
  for (int m = 16; m >= 1; m >>= 1) ss += __shfl_xor(ss, m, 32);

  const float scale = 1.f / fmaxf(sqrtf(ss), 1e-8f);

  v8h o0, o1;
#pragma unroll
  for (int i = 0; i < 8; ++i) o0[i] = (_Float16)(v[i] * scale);
#pragma unroll
  for (int i = 0; i < 8; ++i) o1[i] = (_Float16)(v[8 + i] * scale);

  _Float16* orow = xn + (size_t)row * D_DIM + lane * 16;
  *(v8h*)orow       = o0;
  *((v8h*)orow + 1) = o1;
}

// ---------------------------------------------------------------------------
// Kernel 2: per batch, out = exp(-0.5 * xn @ xn^T).
// 128x128 tile per block (8 waves). Double-buffered LDS fed by
// GLOBAL_LOAD_ASYNC_TO_LDS_B128 (ASYNCcnt), WMMA f32_16x16x32_f16 core.
// ---------------------------------------------------------------------------
__global__ __launch_bounds__(256) void gemm_exp_kernel(
    const _Float16* __restrict__ xn, float* __restrict__ out) {
  // [2 buffers][A tile | B tile]
  __shared__ _Float16 smem[2 * 2 * TILE_HALVES];

  const int b     = blockIdx.z;
  const int mBase = blockIdx.y * TILE;
  const int nBase = blockIdx.x * TILE;

  const _Float16* xb = xn + (size_t)b * N_DIM * D_DIM;
  float*          ob = out + (size_t)b * N_DIM * N_DIM;

  const int tid  = threadIdx.x;
  const int lane = tid & 31;
  const int wave = tid >> 5;
  const int wm   = (wave >> 1) * 32;  // 4 waves stacked in M
  const int wn   = (wave & 1) * 64;   // 2 waves in N

  v8f acc[2][4];
#pragma unroll
  for (int i = 0; i < 2; ++i)
#pragma unroll
    for (int j = 0; j < 4; ++j) acc[i][j] = {};

  // ISA 16-bit A-fragment mapping: lane%16 = row-in-tile, lane/16 selects K octet
  const int fr = lane & 15;
  const int fk = (lane >> 4) * 8;

  // global->LDS staging: 2 threads per row, 32 halves (64 B) each
  const int lrow  = tid >> 1;
  const int lpart = (tid & 1) * 32;

  // Per-thread global source pointers (k-chunk offset added per issue)
  const _Float16* gA0 = xb + (size_t)(mBase + lrow) * D_DIM + lpart;
  const _Float16* gB0 = xb + (size_t)(nBase + lrow) * D_DIM + lpart;

  // Per-thread LDS destination byte offsets (flat addr truncates to LDS addr)
  const unsigned sThread = (unsigned)(size_t)(smem + lrow * LDK + lpart);

  // Issue one chunk's async copies: 8 x B128 per thread (4 A + 4 B)
  auto issue_chunk = [&](int kc0, int buf) {
    const _Float16* gA = gA0 + kc0;
    const _Float16* gB = gB0 + kc0;
    const unsigned sA = sThread + (unsigned)(buf * 2 * TILE_HALVES) * 2u;
    const unsigned sB = sA + (unsigned)TILE_HALVES * 2u;
#pragma unroll
    for (int i = 0; i < 4; ++i) {
      asm volatile("global_load_async_to_lds_b128 %0, %1, off"
                   :: "v"(sA + (unsigned)(i * 16)), "v"(gA + i * 8)
                   : "memory");
      asm volatile("global_load_async_to_lds_b128 %0, %1, off"
                   :: "v"(sB + (unsigned)(i * 16)), "v"(gB + i * 8)
                   : "memory");
    }
  };

  // Prologue: start chunk 0 into buffer 0
  issue_chunk(0, 0);

  for (int ci = 0; ci < NCHUNK; ++ci) {
    const int cur = ci & 1;
    if (ci + 1 < NCHUNK) {
      issue_chunk((ci + 1) * KC, cur ^ 1);       // prefetch next chunk
      asm volatile("s_wait_asynccnt 0x8" ::: "memory");  // chunk ci done (ours)
    } else {
      asm volatile("s_wait_asynccnt 0x0" ::: "memory");
    }
    __syncthreads();  // chunk ci visible from all waves

    const _Float16* sAbuf = smem + cur * 2 * TILE_HALVES;
    const _Float16* sBbuf = sAbuf + TILE_HALVES;

#pragma unroll
    for (int kk = 0; kk < KC; kk += 32) {
      v16h afr[2], bfr[4];
#pragma unroll
      for (int i = 0; i < 2; ++i) {
        const _Float16* p = sAbuf + (wm + i * 16 + fr) * LDK + kk + fk;
        v8h lo = *(const v8h*)p;
        v8h hi = *(const v8h*)(p + 16);
        afr[i] = __builtin_shufflevector(lo, hi, 0, 1, 2, 3, 4, 5, 6, 7,
                                         8, 9, 10, 11, 12, 13, 14, 15);
      }
#pragma unroll
      for (int j = 0; j < 4; ++j) {
        const _Float16* p = sBbuf + (wn + j * 16 + fr) * LDK + kk + fk;
        v8h lo = *(const v8h*)p;
        v8h hi = *(const v8h*)(p + 16);
        bfr[j] = __builtin_shufflevector(lo, hi, 0, 1, 2, 3, 4, 5, 6, 7,
                                         8, 9, 10, 11, 12, 13, 14, 15);
      }
#pragma unroll
      for (int i = 0; i < 2; ++i)
#pragma unroll
        for (int j = 0; j < 4; ++j)
          acc[i][j] = __builtin_amdgcn_wmma_f32_16x16x32_f16(
              false, afr[i], false, bfr[j], (short)0, acc[i][j], false, false);
    }

    __syncthreads();  // all waves done with buffer `cur` before it is refilled
  }

  // Epilogue: C layout = VGPR v, lanes 0-15 -> M=v, lanes 16-31 -> M=8+v; N=lane%16
  const int mHalf = (lane >> 4) * 8;
  const int nCol  = lane & 15;
#pragma unroll
  for (int i = 0; i < 2; ++i) {
#pragma unroll
    for (int j = 0; j < 4; ++j) {
      const int row0 = mBase + wm + i * 16 + mHalf;
      const int col  = nBase + wn + j * 16 + nCol;
#pragma unroll
      for (int v = 0; v < 8; ++v) {
        float s = acc[i][j][v];
        ob[(size_t)(row0 + v) * N_DIM + col] = __expf(-0.5f * s);
      }
    }
  }
}

// ---------------------------------------------------------------------------
extern "C" void kernel_launch(void* const* d_in, const int* in_sizes, int n_in,
                              void* d_out, int out_size, void* d_ws, size_t ws_size,
                              hipStream_t stream) {
  const float* x   = (const float*)d_in[0];
  float*       out = (float*)d_out;
  _Float16*    xn  = (_Float16*)d_ws;  // 4*4096*512 f16 = 16 MB scratch

  normalize_kernel<<<(4 * N_DIM) / 8, 256, 0, stream>>>(x, xn);

  dim3 grid(N_DIM / TILE, N_DIM / TILE, 4);  // 32 x 32 x 4
  gemm_exp_kernel<<<grid, 256, 0, stream>>>(xn, out);
}
